// MambaMixer_16982300688671
// MI455X (gfx1250) — matrix operand order repaired and checked by reading, not compiled, not run on hardware
//
#include <hip/hip_runtime.h>

#define BM 128
#define BN 256
#define BK 32
#define NT 256

typedef __attribute__((ext_vector_type(16))) __bf16 v16bf;
typedef __attribute__((ext_vector_type(4)))  __bf16 v4bf;
typedef __attribute__((ext_vector_type(8)))  float  v8f;

__device__ __forceinline__ float silu_f(float v) {
    return v / (1.0f + __expf(-v));
}

// permute K index for the 16-bit WMMA A-fragment half layout:
// stored order [0..7, 16..23, 8..15, 24..31] -> lanes read 16 contiguous bf16
__device__ __forceinline__ int kswz(int k) {
    return k ^ ((((k >> 3) ^ (k >> 4)) & 1) ? 24 : 0);
}

__device__ __forceinline__ v4bf pack4(float4 v) {
    v4bf o;
    o[0] = (__bf16)v.x; o[1] = (__bf16)v.y; o[2] = (__bf16)v.z; o[3] = (__bf16)v.w;
    return o;
}

// -------------------------------------------------------------------------
// 128x256 block / 64x64 wave-tile WMMA GEMM: C[M,N] = A[M,K] @ B[K,N]
// f32 global operands, bf16 WMMA compute, f32 accumulate/out.
// Requires K % 32 == 0, M % 128 == 0. N may be ragged (guarded).
// 8 waves: wm = wave&1 (2 x 64 rows), wn = wave>>1 (4 x 64 cols).
// Ping-pong LDS, one barrier per K-step, 16 WMMA : 16 ds_load_b128 per wave.
// EPI==0: plain store. EPI==1: v = clip(softplus(v + bias[n]), 1e-3, 0.1)
// -------------------------------------------------------------------------
template<int EPI>
__global__ __launch_bounds__(NT)
void gemm128(const float* __restrict__ A, int lda,
             const float* __restrict__ B, int ldb,
             float*       __restrict__ C, int ldc,
             int M, int N, int K,
             const float* __restrict__ bias)
{
    __shared__ __align__(32) __bf16 As[2][BM * BK];
    __shared__ __align__(32) __bf16 Bs[2][BN * BK];

    const int tid  = threadIdx.x;
    const int lane = tid & 31;
    const int wave = tid >> 5;
    const int wm   = wave & 1;
    const int wn   = wave >> 1;
    const int ll   = lane & 15;
    const int lh   = lane >> 4;

    const int n0 = blockIdx.x * BN;
    const int m0 = blockIdx.y * BM;

    auto stage = [&](int buf, int k0) {
        // A: 4-float chunks, coalesced, swizzled ds_store_b64
        #pragma unroll
        for (int j = 0; j < (BM * BK) / (NT * 4); ++j) {
            int c   = tid + j * NT;
            int row = c >> 3;
            int kc  = (c & 7) << 2;
            float4 v = *reinterpret_cast<const float4*>(
                &A[(size_t)(m0 + row) * lda + k0 + kc]);
            if (j == 0 && k0 + BK < K)
                __builtin_prefetch(&A[(size_t)(m0 + row) * lda + k0 + BK + kc], 0, 1);
            *reinterpret_cast<v4bf*>(&As[buf][row * BK + kswz(kc)]) = pack4(v);
        }
        // B: transpose-stage Bs[n][k]; 4 strided b32 loads -> one ds_store_b64
        #pragma unroll
        for (int j = 0; j < (BN * BK) / (NT * 4); ++j) {
            int c  = tid + j * NT;
            int n  = c & (BN - 1);
            int kc = (c >> 8) << 2;
            int gn = n0 + n;
            float msk = (gn < N) ? 1.0f : 0.0f;
            int  gc   = (gn < N) ? gn : (N - 1);
            const float* bp = &B[(size_t)(k0 + kc) * ldb + gc];
            if (j == 0 && k0 + BK < K)
                __builtin_prefetch(&B[(size_t)(k0 + BK + kc) * ldb + gc], 0, 1);
            float4 v;
            v.x = bp[0] * msk;
            v.y = bp[(size_t)ldb] * msk;
            v.z = bp[(size_t)ldb * 2] * msk;
            v.w = bp[(size_t)ldb * 3] * msk;
            *reinterpret_cast<v4bf*>(&Bs[buf][n * BK + kswz(kc)]) = pack4(v);
        }
    };

    v8f acc[4][4] = {};

    stage(0, 0);
    __syncthreads();

    const int nk = K / BK;
    for (int kb = 0; kb < nk; ++kb) {
        const int p = kb & 1;
        if (kb + 1 < nk) stage(p ^ 1, (kb + 1) * BK);

        v16bf af[4];
        #pragma unroll
        for (int tm = 0; tm < 4; ++tm) {
            int row = wm * 64 + tm * 16 + ll;
            af[tm] = *reinterpret_cast<const v16bf*>(&As[p][row * BK + lh * 16]);
        }
        #pragma unroll
        for (int tn = 0; tn < 4; ++tn) {
            int col = wn * 64 + tn * 16 + ll;
            v16bf bfr = *reinterpret_cast<const v16bf*>(&Bs[p][col * BK + lh * 16]);
            #pragma unroll
            for (int tm = 0; tm < 4; ++tm)
                acc[tm][tn] = __builtin_amdgcn_wmma_f32_16x16x32_bf16(
                    false, af[tm], false, bfr, (short)0, acc[tm][tn], false, false);
        }
        __syncthreads();
    }

    #pragma unroll
    for (int tm = 0; tm < 4; ++tm) {
        #pragma unroll
        for (int tn = 0; tn < 4; ++tn) {
            int col = n0 + wn * 64 + tn * 16 + ll;
            if (col >= N) continue;
            int base = m0 + wm * 64 + tm * 16 + lh * 8;
            #pragma unroll
            for (int r = 0; r < 8; ++r) {
                float v = acc[tm][tn][r];
                if (EPI == 1) {
                    v += bias[col];
                    float sp = (v > 20.0f) ? v : log1pf(__expf(v));
                    v = fminf(fmaxf(sp, 0.001f), 0.1f);
                }
                C[(size_t)(base + r) * ldc + col] = v;
            }
        }
    }
}

// -------------------------------------------------------------------------
// Grouped causal conv1d (D_CONV=4, 2 groups) as a 128-step accumulated
// WMMA GEMM (4 taps x 32 K-chunks), fused bias + SiLU -> u.
// xs = xz[:, :, 0:2048] (row stride 4096). Branchless causal zero-pad.
// Same 128x256 / 64x64 wave tiling as gemm128.
// -------------------------------------------------------------------------
__global__ __launch_bounds__(NT)
void conv_silu_gemm(const float* __restrict__ xz,
                    const float* __restrict__ ck,
                    const float* __restrict__ cb,
                    float* __restrict__ u)
{
    __shared__ __align__(32) __bf16 As[2][BM * BK];
    __shared__ __align__(32) __bf16 Bs[2][BN * BK];

    const int tid  = threadIdx.x;
    const int lane = tid & 31;
    const int wave = tid >> 5;
    const int wm   = wave & 1;
    const int wn   = wave >> 1;
    const int ll   = lane & 15;
    const int lh   = lane >> 4;

    const int n0  = blockIdx.x * BN;        // output channel tile
    const int m0  = blockIdx.y * BM;        // (b,t) row tile
    const int grp = (n0 >= 1024) ? 1 : 0;   // feature group (256 | 1024)

    auto stage = [&](int buf, int s) {
        const int w  = s >> 5;          // tap 0..3
        const int k0 = (s & 31) << 5;   // input-channel chunk
        #pragma unroll
        for (int j = 0; j < (BM * BK) / (NT * 4); ++j) {
            int c   = tid + j * NT;
            int row = c >> 3;
            int kc  = (c & 7) << 2;
            int gr  = m0 + row;
            int b   = gr >> 11;
            int t   = gr & 2047;
            int tt  = t + w - 3;                    // causal shift
            int tc  = (tt < 0) ? 0 : tt;            // clamped (branchless)
            float m = (tt < 0) ? 0.0f : 1.0f;
            float4 v = *reinterpret_cast<const float4*>(
                &xz[((size_t)((b << 11) + tc)) * 4096 + grp * 1024 + k0 + kc]);
            v.x *= m; v.y *= m; v.z *= m; v.w *= m;
            *reinterpret_cast<v4bf*>(&As[buf][row * BK + kswz(kc)]) = pack4(v);
        }
        #pragma unroll
        for (int j = 0; j < (BN * BK) / (NT * 4); ++j) {
            int c  = tid + j * NT;
            int n  = c & (BN - 1);
            int kc = (c >> 8) << 2;
            const float* bp = &ck[(size_t)w * (1024 * 2048) +
                                  (size_t)(k0 + kc) * 2048 + (n0 + n)];
            float4 v;
            v.x = bp[0];
            v.y = bp[2048];
            v.z = bp[4096];
            v.w = bp[6144];
            *reinterpret_cast<v4bf*>(&Bs[buf][n * BK + kswz(kc)]) = pack4(v);
        }
    };

    v8f acc[4][4] = {};

    const int NSTEP = 4 * (1024 / BK);   // 128
    stage(0, 0);
    __syncthreads();

    for (int s = 0; s < NSTEP; ++s) {
        const int p = s & 1;
        if (s + 1 < NSTEP) stage(p ^ 1, s + 1);

        v16bf af[4];
        #pragma unroll
        for (int tm = 0; tm < 4; ++tm) {
            int row = wm * 64 + tm * 16 + ll;
            af[tm] = *reinterpret_cast<const v16bf*>(&As[p][row * BK + lh * 16]);
        }
        #pragma unroll
        for (int tn = 0; tn < 4; ++tn) {
            int col = wn * 64 + tn * 16 + ll;
            v16bf bfr = *reinterpret_cast<const v16bf*>(&Bs[p][col * BK + lh * 16]);
            #pragma unroll
            for (int tm = 0; tm < 4; ++tm)
                acc[tm][tn] = __builtin_amdgcn_wmma_f32_16x16x32_bf16(
                    false, af[tm], false, bfr, (short)0, acc[tm][tn], false, false);
        }
        __syncthreads();
    }

    #pragma unroll
    for (int tm = 0; tm < 4; ++tm) {
        #pragma unroll
        for (int tn = 0; tn < 4; ++tn) {
            int col  = n0 + wn * 64 + tn * 16 + ll;
            int base = m0 + wm * 64 + tm * 16 + lh * 8;
            #pragma unroll
            for (int r = 0; r < 8; ++r) {
                float v = acc[tm][tn][r] + cb[col];
                u[(size_t)(base + r) * 2048 + col] = silu_f(v);
            }
        }
    }
}

// -------------------------------------------------------------------------
// Selective scan: thread = (b, d); h[0..15] and A-row in registers.
// B_t / C_t broadcast through double-buffered LDS, read as float4s.
// Fused: y = scan(t) + u*D ; g = y * silu(res)  (res = xz[:, :, 2048+d])
// -------------------------------------------------------------------------
__global__ __launch_bounds__(NT)
void scan_kernel(const float* __restrict__ dbc,
                 const float* __restrict__ delta,
                 const float* __restrict__ u,
                 const float* __restrict__ xz,
                 const float* __restrict__ A_log,
                 const float* __restrict__ Dp,
                 float* __restrict__ g)
{
    const int b = blockIdx.x >> 3;
    const int d = (blockIdx.x & 7) * NT + threadIdx.x;

    float a[16], h[16];
    #pragma unroll
    for (int n = 0; n < 16; ++n) {
        a[n] = -__expf(A_log[d * 16 + n]);
        h[n] = 0.0f;
    }
    const float Dd = Dp[d];

    __shared__ __align__(16) float sBC[2][32];

    for (int t = 0; t < 2048; ++t) {
        const int p = t & 1;
        const size_t row = ((size_t)b << 11) + t;
        if (threadIdx.x < 32)
            sBC[p][threadIdx.x] = dbc[row * 96 + 64 + threadIdx.x];
        __syncthreads();

        float dv = delta[row * 2048 + d];
        float uv = u[row * 2048 + d];
        float rv = xz[row * 4096 + 2048 + d];
        float du = dv * uv;
        float y  = 0.0f;
        #pragma unroll
        for (int q = 0; q < 4; ++q) {
            float4 B4 = *reinterpret_cast<const float4*>(&sBC[p][q * 4]);
            float4 C4 = *reinterpret_cast<const float4*>(&sBC[p][16 + q * 4]);
            float Bl[4] = {B4.x, B4.y, B4.z, B4.w};
            float Cl[4] = {C4.x, C4.y, C4.z, C4.w};
            #pragma unroll
            for (int e = 0; e < 4; ++e) {
                int n = q * 4 + e;
                float dA = __expf(dv * a[n]);
                h[n] = dA * h[n] + du * Bl[e];
                y += h[n] * Cl[e];
            }
        }
        float yt = y + uv * Dd;
        g[row * 2048 + d] = yt * silu_f(rv);
    }
}

// -------------------------------------------------------------------------
extern "C" void kernel_launch(void* const* d_in, const int* in_sizes, int n_in,
                              void* d_out, int out_size, void* d_ws, size_t ws_size,
                              hipStream_t stream) {
    const float* x      = (const float*)d_in[0];
    const float* W_in   = (const float*)d_in[1];
    const float* conv_k = (const float*)d_in[2];
    const float* conv_b = (const float*)d_in[3];
    const float* W_x    = (const float*)d_in[4];
    const float* W_dt   = (const float*)d_in[5];
    const float* b_dt   = (const float*)d_in[6];
    const float* A_log  = (const float*)d_in[7];
    const float* Dp     = (const float*)d_in[8];
    const float* W_out  = (const float*)d_in[9];
    float* out = (float*)d_out;

    float* ws    = (float*)d_ws;
    float* xz    = ws;                       // 4096 x 4096
    float* u     = xz    + 16777216;         // 4096 x 2048
    float* dbc   = u     + 8388608;          // 4096 x 96
    float* delta = dbc   + 393216;           // 4096 x 2048
    float* gbuf  = delta + 8388608;          // 4096 x 2048

    dim3 blk(NT);

    // 1) xz = x @ W_in            M=4096 N=4096 K=1024
    gemm128<0><<<dim3(16, 32), blk, 0, stream>>>(x, 1024, W_in, 4096, xz, 4096,
                                                 4096, 4096, 1024, nullptr);
    // 2) u = silu(conv(xs) + cb)  (4-tap grouped GEMM)
    conv_silu_gemm<<<dim3(8, 32), blk, 0, stream>>>(xz, conv_k, conv_b, u);
    // 3) dbc = u @ W_x            M=4096 N=96 K=2048
    gemm128<0><<<dim3(1, 32), blk, 0, stream>>>(u, 2048, W_x, 96, dbc, 96,
                                                4096, 96, 2048, nullptr);
    // 4) delta = clip(softplus(dbc[:, :64] @ W_dt + b_dt))  M=4096 N=2048 K=64
    gemm128<1><<<dim3(8, 32), blk, 0, stream>>>(dbc, 96, W_dt, 2048, delta, 2048,
                                                 4096, 2048, 64, b_dt);
    // 5) selective scan -> g (fused +u*D and *silu(res))
    scan_kernel<<<dim3(16), blk, 0, stream>>>(dbc, delta, u, xz, A_log, Dp, gbuf);
    // 6) out = g @ W_out          M=4096 N=1024 K=2048
    gemm128<0><<<dim3(4, 32), blk, 0, stream>>>(gbuf, 2048, W_out, 1024, out, 1024,
                                                4096, 1024, 2048, nullptr);
}
